// RetNetBlock_13606456394499
// MI455X (gfx1250) — matrix operand order repaired and checked by reading, not compiled, not run on hardware
//
#include <hip/hip_runtime.h>
#include <stdint.h>

// ---------------- types ----------------
typedef __attribute__((ext_vector_type(16))) __bf16 v16bf;
typedef __attribute__((ext_vector_type(8)))  float  v8f;

struct alignas(16) U4 { unsigned int x, y, z, w; };
union Frag16 { U4 q[2]; v16bf v; };

#define HIDDEN  1024
#define NHEADS  16
#define HD      64
#define BATCH   2
#define SEQ     2048
#define NROWS   (BATCH*SEQ)   // 4096
#define ROT     32

static __device__ __forceinline__ uint16_t f2bf(float f) {
  union { float f; uint32_t u; } v; v.f = f;
  uint32_t u = v.u;
  return (uint16_t)((u + 0x7FFFu + ((u >> 16) & 1u)) >> 16);
}

static __device__ __forceinline__ v8f wmma_bf16(v16bf a, v16bf b, v8f c) {
  return __builtin_amdgcn_wmma_f32_16x16x32_bf16(false, a, false, b, (short)0, c,
                                                 false, false);
}

// CDNA5 async global->LDS copy, 16B per lane, tracked by ASYNCcnt.
static __device__ __forceinline__ void async_copy_b128(void* lds_ptr,
                                                       const void* gaddr) {
  uint32_t lds_off = (uint32_t)(uintptr_t)lds_ptr;          // LDS aperture low bits
  asm volatile("global_load_async_to_lds_b128 %0, %1, off"
               :: "v"(lds_off), "v"((unsigned long long)(uintptr_t)gaddr)
               : "memory");
}
#define WAIT_ASYNC_LE(n) asm volatile("s_wait_asynccnt %0" :: "n"(n) : "memory")

// ---------------- elementwise / norm kernels ----------------
// fused fp32 -> bf16 transpose: src is KxN, dst is NxK
__global__ __launch_bounds__(256)
void k_transpose_bf16(const float* __restrict__ src, uint16_t* __restrict__ dst,
                      int K, int N) {
  __shared__ uint16_t t[32][33];
  int nb = blockIdx.x * 32, kb = blockIdx.y * 32;
  int tx = threadIdx.x & 31, ty = threadIdx.x >> 5;   // 32 x 8
#pragma unroll
  for (int i = 0; i < 32; i += 8)
    t[ty + i][tx] = f2bf(src[(size_t)(kb + ty + i) * N + nb + tx]);
  __syncthreads();
#pragma unroll
  for (int i = 0; i < 32; i += 8)
    dst[(size_t)(nb + ty + i) * K + kb + tx] = t[tx][ty + i];
}

// one block per row (1024 cols), 256 threads
__global__ __launch_bounds__(256)
void k_layernorm_bf16(const float* __restrict__ x, const float* __restrict__ w,
                      const float* __restrict__ b, uint16_t* __restrict__ out) {
  int row = blockIdx.x;
  const float* xr = x + (size_t)row * HIDDEN;
  float v[4], s = 0.f, s2 = 0.f;
#pragma unroll
  for (int i = 0; i < 4; ++i) {
    v[i] = xr[threadIdx.x + i * 256];
    s += v[i]; s2 += v[i] * v[i];
  }
  __shared__ float red0[256], red1[256];
  red0[threadIdx.x] = s; red1[threadIdx.x] = s2;
  __syncthreads();
  for (int off = 128; off > 0; off >>= 1) {
    if (threadIdx.x < off) {
      red0[threadIdx.x] += red0[threadIdx.x + off];
      red1[threadIdx.x] += red1[threadIdx.x + off];
    }
    __syncthreads();
  }
  float mean = red0[0] * (1.0f / HIDDEN);
  float var  = red1[0] * (1.0f / HIDDEN) - mean * mean;
  float rinv = rsqrtf(var + 1e-5f);
#pragma unroll
  for (int i = 0; i < 4; ++i) {
    int c = threadIdx.x + i * 256;
    out[(size_t)row * HIDDEN + c] = f2bf((v[i] - mean) * rinv * w[c] + b[c]);
  }
}

__global__ void k_mish_mul(const float* __restrict__ g,
                           const float* __restrict__ retn,
                           uint16_t* __restrict__ yb, int n) {
  int i = blockIdx.x * blockDim.x + threadIdx.x;
  if (i < n) {
    float x = g[i];
    float sp = (x > 20.f) ? x : log1pf(__expf(x));
    yb[i] = f2bf(x * tanhf(sp) * retn[i]);
  }
}

__global__ void k_residual_inplace(float* __restrict__ o,
                                   const float* __restrict__ x, int n) {
  int i = blockIdx.x * blockDim.x + threadIdx.x;
  if (i < n) o[i] += x[i];
}

__global__ void k_gelu_bf16(const float* __restrict__ f1,
                            uint16_t* __restrict__ f1g, int n) {
  int i = blockIdx.x * blockDim.x + threadIdx.x;
  if (i < n) {
    float x = f1[i];
    f1g[i] = f2bf(0.5f * x * (1.0f + erff(x * 0.70710678118654752f)));
  }
}

__global__ void k_add_out(const float* __restrict__ f2,
                          const float* __restrict__ y2,
                          float* __restrict__ out, int n) {
  int i = blockIdx.x * blockDim.x + threadIdx.x;
  if (i < n) out[i] = f2[i] + y2[i];
}

// ---------------- GEMM: C(MxN) = A(MxK,bf16) @ Bt(NxK,bf16)^T [+bias] --------
// Both operands row-contiguous in K -> straight async b128 staging, double buffer
#define BM 128
#define BN 128
#define BK 32

__global__ __launch_bounds__(256)
void k_gemm_bf16(const uint16_t* __restrict__ A, const uint16_t* __restrict__ Bt,
                 const float* __restrict__ bias, float* __restrict__ C,
                 int M, int N, int K) {
  __shared__ alignas(16) uint16_t lsA[2][BM][BK];   // 16KB
  __shared__ alignas(16) uint16_t lsB[2][BN][BK];   // 16KB, [n][k]
  int tid  = threadIdx.x;
  int wave = tid >> 5, lane = tid & 31;
  int wm = wave >> 1, wn = wave & 1;
  int l15 = lane & 15, l16 = lane >> 4;
  int mBlk = blockIdx.y * BM, nBlk = blockIdx.x * BN;

  v8f zacc = {0.f,0.f,0.f,0.f,0.f,0.f,0.f,0.f};
  v8f acc[2][4];
#pragma unroll
  for (int i = 0; i < 2; ++i)
#pragma unroll
    for (int j = 0; j < 4; ++j) acc[i][j] = zacc;

  int srow = tid >> 2;            // 0..63
  int sseg = (tid & 3) * 8;       // element offset within 32-elem row

  auto stage = [&](int buf, int k0) {
    async_copy_b128(&lsA[buf][srow][sseg],
                    A + (size_t)(mBlk + srow) * K + k0 + sseg);
    async_copy_b128(&lsA[buf][64 + srow][sseg],
                    A + (size_t)(mBlk + 64 + srow) * K + k0 + sseg);
    async_copy_b128(&lsB[buf][srow][sseg],
                    Bt + (size_t)(nBlk + srow) * K + k0 + sseg);
    async_copy_b128(&lsB[buf][64 + srow][sseg],
                    Bt + (size_t)(nBlk + 64 + srow) * K + k0 + sseg);
  };

  stage(0, 0);
  int buf = 0;
  for (int k0 = 0; k0 < K; k0 += BK) {
    bool hasNext = (k0 + BK) < K;
    if (hasNext) {
      stage(buf ^ 1, k0 + BK);          // prefetch next tile (async, in-flight)
      WAIT_ASYNC_LE(4);                 // current tile's 4 wave-ops complete
    } else {
      WAIT_ASYNC_LE(0);
    }
    __syncthreads();

    Frag16 fa[2], fb[4];
#pragma unroll
    for (int mf = 0; mf < 2; ++mf) {
      const uint16_t* p = &lsA[buf][wm * 32 + mf * 16 + l15][l16 * 8];
      fa[mf].q[0] = *(const U4*)p;
      fa[mf].q[1] = *(const U4*)(p + 16);    // K += 16
    }
#pragma unroll
    for (int nf = 0; nf < 4; ++nf) {
      const uint16_t* p = &lsB[buf][wn * 64 + nf * 16 + l15][l16 * 8];
      fb[nf].q[0] = *(const U4*)p;
      fb[nf].q[1] = *(const U4*)(p + 16);
    }
#pragma unroll
    for (int mf = 0; mf < 2; ++mf)
#pragma unroll
      for (int nf = 0; nf < 4; ++nf)
        acc[mf][nf] = wmma_bf16(fa[mf].v, fb[nf].v, acc[mf][nf]);
    __syncthreads();
    buf ^= 1;
  }

#pragma unroll
  for (int mf = 0; mf < 2; ++mf) {
    int mrow0 = mBlk + wm * 32 + mf * 16 + l16 * 8;  // C layout: M = r + 8*(lane>=16)
#pragma unroll
    for (int nf = 0; nf < 4; ++nf) {
      int ncol = nBlk + wn * 64 + nf * 16 + l15;     // N = lane&15
      float bv = bias ? bias[ncol] : 0.0f;
#pragma unroll
      for (int r = 0; r < 8; ++r)
        C[(size_t)(mrow0 + r) * N + ncol] = acc[mf][nf][r] + bv;
    }
  }
}

// ---------------- xPos rotary + head split/pack ----------------
__global__ __launch_bounds__(64)
void k_rotary(const float* __restrict__ qkv, uint16_t* __restrict__ qb,
              uint16_t* __restrict__ kb, uint16_t* __restrict__ vt) {
  int bid = blockIdx.x;
  int h = bid % NHEADS;
  int s = (bid / NHEADS) % SEQ;
  int b = bid / (NHEADS * SEQ);
  int d = threadIdx.x;
  const float* base = qkv + (size_t)(b * SEQ + s) * (3 * HIDDEN) + h * (3 * HD);
  float q = base[d], k = base[HD + d], v = base[2 * HD + d];
  float qo = q, ko = k;
  if (d < ROT) {
    int p = d >> 1;
    float invf = __powf(10000.0f, -(float)(2 * p) / (float)ROT);
    float ang = (float)s * invf;
    float sn, c;
    __sincosf(ang, &sn, &c);
    float sval = ((float)(2 * p) + 0.4f * (float)ROT) / (1.4f * (float)ROT);
    float power = ((float)s - (float)(SEQ / 2)) * (1.0f / 512.0f);
    float sc = __powf(sval, power);
    float rsc = 1.0f / sc;
    float qpart = (d & 1) ? base[d - 1] : base[d + 1];
    float kpart = (d & 1) ? base[HD + d - 1] : base[HD + d + 1];
    float rh_q = (d & 1) ? qpart : -qpart;   // rotate_half
    float rh_k = (d & 1) ? kpart : -kpart;
    qo = q * c * sc  + rh_q * sn * sc;
    ko = k * c * rsc + rh_k * sn * rsc;
  }
  size_t bh = (size_t)(b * NHEADS + h);
  qb[(bh * SEQ + s) * HD + d] = f2bf(qo);
  kb[(bh * SEQ + s) * HD + d] = f2bf(ko);
  vt[(bh * HD + d) * SEQ + s] = f2bf(v);     // V stored transposed [d][t]
}

// ---------------- retention: out = (QK^T * decay) V, causal ----------------
// grid: (BATCH*NHEADS)*(SEQ/64) blocks of 128 threads (4 waves, 16 s-rows each)
#define ST 64
__global__ __launch_bounds__(128)
void k_retention(const uint16_t* __restrict__ qb, const uint16_t* __restrict__ kb,
                 const uint16_t* __restrict__ vt, float* __restrict__ ret) {
  __shared__ alignas(16) uint16_t lsK[2][ST][HD];   // 16KB  [t][d]
  __shared__ alignas(16) uint16_t lsV[2][HD][ST];   // 16KB  [d][t]
  __shared__ alignas(16) uint16_t lsS[4][16][ST];   // 8KB   per-wave scores
  const int nChunks = SEQ / ST;                      // 32
  int chunk = blockIdx.x % nChunks;
  int bh = blockIdx.x / nChunks;
  int h = bh % NHEADS;
  int tid = threadIdx.x, wave = tid >> 5, lane = tid & 31;
  int l15 = lane & 15, l16 = lane >> 4;

  float gamma = 1.0f - exp2f(-5.0f - (float)h);
  float lg = log2f(gamma);

  const uint16_t* qB = qb + (size_t)bh * SEQ * HD;
  const uint16_t* kB = kb + (size_t)bh * SEQ * HD;
  const uint16_t* vB = vt + (size_t)bh * HD * SEQ;

  int sWave = chunk * ST + wave * 16;

  Frag16 fq[2];           // Q rows held in registers across the whole t-loop
#pragma unroll
  for (int ks = 0; ks < 2; ++ks) {
    const uint16_t* p = qB + (size_t)(sWave + l15) * HD + ks * 32 + l16 * 8;
    fq[ks].q[0] = *(const U4*)p;
    fq[ks].q[1] = *(const U4*)(p + 16);
  }

  v8f zacc = {0.f,0.f,0.f,0.f,0.f,0.f,0.f,0.f};
  v8f oacc[4];
#pragma unroll
  for (int i = 0; i < 4; ++i) oacc[i] = zacc;

  int rrow = tid >> 1;              // 0..63
  int rseg = (tid & 1) * 32;        // element base within 64-elem row

  auto stageKV = [&](int buf, int tBase) {
#pragma unroll
    for (int i = 0; i < 4; ++i) {
      int e = rseg + i * 8;
      async_copy_b128(&lsK[buf][rrow][e],
                      kB + (size_t)(tBase + rrow) * HD + e);
    }
#pragma unroll
    for (int i = 0; i < 4; ++i) {
      int e = rseg + i * 8;
      async_copy_b128(&lsV[buf][rrow][e],
                      vB + (size_t)rrow * SEQ + tBase + e);
    }
  };

  stageKV(0, 0);
  int buf = 0;
  for (int tt = 0; tt <= chunk; ++tt) {
    int tBase = tt * ST;
    bool hasNext = tt < chunk;
    if (hasNext) {
      stageKV(buf ^ 1, tBase + ST);   // stream next K/V tiles while computing
      WAIT_ASYNC_LE(8);
    } else {
      WAIT_ASYNC_LE(0);
    }
    __syncthreads();

    // ---- scores = Q K^T  (B operand: lane=col t, d as two 8-runs, 16 apart) ----
    v8f sacc[4];
#pragma unroll
    for (int i = 0; i < 4; ++i) sacc[i] = zacc;
#pragma unroll
    for (int ks = 0; ks < 2; ++ks) {
#pragma unroll
      for (int nf = 0; nf < 4; ++nf) {
        Frag16 fk;
        const uint16_t* p = &lsK[buf][nf * 16 + l15][ks * 32 + l16 * 8];
        fk.q[0] = *(const U4*)p;
        fk.q[1] = *(const U4*)(p + 16);
        sacc[nf] = wmma_bf16(fq[ks].v, fk.v, sacc[nf]);
      }
    }

    // ---- decay scale + bf16, route D-layout -> A-layout through LDS ----
#pragma unroll
    for (int nf = 0; nf < 4; ++nf) {
      int tg = tBase + nf * 16 + l15;
#pragma unroll
      for (int r = 0; r < 8; ++r) {
        int m = r + l16 * 8;
        int delta = (sWave + m) - tg;
        float val = (delta >= 0) ? sacc[nf][r] * exp2f((float)delta * lg) : 0.0f;
        lsS[wave][m][nf * 16 + l15] = f2bf(val);
      }
    }
    // per-wave LDS region: in-wave RAW, LDS is in-order + dscnt wait

    Frag16 fs[2];
#pragma unroll
    for (int ks2 = 0; ks2 < 2; ++ks2) {
      const uint16_t* p = &lsS[wave][l15][ks2 * 32 + l16 * 8];
      fs[ks2].q[0] = *(const U4*)p;
      fs[ks2].q[1] = *(const U4*)(p + 16);
    }

    // ---- out += scores @ V  (B operand from V^T: lane=col d, t as two 8-runs, 16 apart) --
#pragma unroll
    for (int ks2 = 0; ks2 < 2; ++ks2) {
#pragma unroll
      for (int df = 0; df < 4; ++df) {
        Frag16 fv;
        const uint16_t* p = &lsV[buf][df * 16 + l15][ks2 * 32 + l16 * 8];
        fv.q[0] = *(const U4*)p;
        fv.q[1] = *(const U4*)(p + 16);
        oacc[df] = wmma_bf16(fs[ks2].v, fv.v, oacc[df]);
      }
    }
    __syncthreads();
    buf ^= 1;
  }

#pragma unroll
  for (int df = 0; df < 4; ++df)
#pragma unroll
    for (int r = 0; r < 8; ++r) {
      int sg = sWave + r + l16 * 8;
      ret[((size_t)bh * SEQ + sg) * HD + df * 16 + l15] = oacc[df][r];
    }
}

// ---------------- per-head groupnorm over (S, D), write transposed ----------
__global__ __launch_bounds__(256)
void k_groupnorm(const float* __restrict__ ret, const float* __restrict__ gn_w,
                 const float* __restrict__ gn_b, float* __restrict__ retn) {
  int bh = blockIdx.x;
  int b = bh / NHEADS, h = bh % NHEADS;
  const float* src = ret + (size_t)bh * SEQ * HD;
  const int NEL = SEQ * HD;
  float s = 0.f, s2 = 0.f;
  for (int i = threadIdx.x; i < NEL; i += 256) {
    float v = src[i];
    s += v; s2 += v * v;
  }
  __shared__ float red0[256], red1[256];
  red0[threadIdx.x] = s; red1[threadIdx.x] = s2;
  __syncthreads();
  for (int off = 128; off > 0; off >>= 1) {
    if (threadIdx.x < off) {
      red0[threadIdx.x] += red0[threadIdx.x + off];
      red1[threadIdx.x] += red1[threadIdx.x + off];
    }
    __syncthreads();
  }
  float mean = red0[0] * (1.0f / NEL);
  float var  = red1[0] * (1.0f / NEL) - mean * mean;
  float rinv = rsqrtf(var + 1e-5f);
  float w = gn_w[h], bb = gn_b[h];
  for (int i = threadIdx.x; i < NEL; i += 256) {
    int sIdx = i / HD, d = i % HD;
    retn[(size_t)(b * SEQ + sIdx) * HIDDEN + h * HD + d] =
        (src[i] - mean) * rinv * w + bb;
  }
}

// ---------------- host orchestration ----------------
extern "C" void kernel_launch(void* const* d_in, const int* in_sizes, int n_in,
                              void* d_out, int out_size, void* d_ws, size_t ws_size,
                              hipStream_t stream) {
  const float* x      = (const float*)d_in[0];
  const float* ln1_w  = (const float*)d_in[1];
  const float* ln1_b  = (const float*)d_in[2];
  const float* wqkv_w = (const float*)d_in[3];
  const float* wg_w   = (const float*)d_in[4];
  const float* wo_w   = (const float*)d_in[5];
  const float* gn_w   = (const float*)d_in[6];
  const float* gn_b   = (const float*)d_in[7];
  const float* ln2_w  = (const float*)d_in[8];
  const float* ln2_b  = (const float*)d_in[9];
  const float* w1_w   = (const float*)d_in[10];
  const float* w1_b   = (const float*)d_in[11];
  const float* w2_w   = (const float*)d_in[12];
  const float* w2_b   = (const float*)d_in[13];
  float* out = (float*)d_out;

  uint8_t* ws = (uint8_t*)d_ws;
  const size_t OFF_WQKVB = 0;                      // [3072][1024] bf16 = 6291456
  const size_t OFF_WGB   = 6291456;                // [1024][1024] bf16
  const size_t OFF_WOB   = 8388608;
  const size_t OFF_W1B   = 10485760;
  const size_t OFF_W2B   = 12582912;
  const size_t OFF_HB    = 14680064;               // 8388608 (later: ZB)
  const size_t OFF_QKV   = 23068672;               // 50331648 (later: RET/RETN/G)
  const size_t OFF_RET   = OFF_QKV;
  const size_t OFF_RETN  = OFF_QKV + 16777216;
  const size_t OFF_G     = OFF_QKV + 33554432;
  const size_t OFF_QB    = 73400320;
  const size_t OFF_KB    = 81788928;
  const size_t OFF_VT    = 90177536;
  const size_t OFF_YB    = 98566144;               // (later: F1G)
  const size_t OFF_Y2    = 106954752;
  const size_t OFF_F1    = 123731968;              // (later: F2)

  uint16_t* wqkvB = (uint16_t*)(ws + OFF_WQKVB);
  uint16_t* wgB   = (uint16_t*)(ws + OFF_WGB);
  uint16_t* woB   = (uint16_t*)(ws + OFF_WOB);
  uint16_t* w1B   = (uint16_t*)(ws + OFF_W1B);
  uint16_t* w2B   = (uint16_t*)(ws + OFF_W2B);
  uint16_t* hb    = (uint16_t*)(ws + OFF_HB);
  float*    qkv   = (float*)   (ws + OFF_QKV);
  float*    retb  = (float*)   (ws + OFF_RET);
  float*    retn  = (float*)   (ws + OFF_RETN);
  float*    gbuf  = (float*)   (ws + OFF_G);
  uint16_t* qb    = (uint16_t*)(ws + OFF_QB);
  uint16_t* kb    = (uint16_t*)(ws + OFF_KB);
  uint16_t* vtb   = (uint16_t*)(ws + OFF_VT);
  uint16_t* yb    = (uint16_t*)(ws + OFF_YB);
  float*    y2    = (float*)   (ws + OFF_Y2);
  float*    f1    = (float*)   (ws + OFF_F1);
  uint16_t* zb    = (uint16_t*)(ws + OFF_HB);   // alias of hb (dead by then)
  uint16_t* f1g   = (uint16_t*)(ws + OFF_YB);   // alias of yb
  float*    f2    = (float*)   (ws + OFF_F1);   // alias of f1

  const int N = NROWS * HIDDEN;  // 4194304
  const int EWG = (N + 255) / 256;

  // 1. weights -> bf16, pre-transposed to [N][K] for async GEMM staging
  k_transpose_bf16<<<dim3(3*HIDDEN/32, HIDDEN/32), 256, 0, stream>>>(wqkv_w, wqkvB, HIDDEN, 3*HIDDEN);
  k_transpose_bf16<<<dim3(HIDDEN/32, HIDDEN/32), 256, 0, stream>>>(wg_w, wgB, HIDDEN, HIDDEN);
  k_transpose_bf16<<<dim3(HIDDEN/32, HIDDEN/32), 256, 0, stream>>>(wo_w, woB, HIDDEN, HIDDEN);
  k_transpose_bf16<<<dim3(HIDDEN/32, HIDDEN/32), 256, 0, stream>>>(w1_w, w1B, HIDDEN, HIDDEN);
  k_transpose_bf16<<<dim3(HIDDEN/32, HIDDEN/32), 256, 0, stream>>>(w2_w, w2B, HIDDEN, HIDDEN);

  // 2. h = LN1(x) -> bf16
  k_layernorm_bf16<<<NROWS, 256, 0, stream>>>(x, ln1_w, ln1_b, hb);

  // 3. qkv = h @ Wqkv
  k_gemm_bf16<<<dim3(3*HIDDEN/BN, NROWS/BM), 256, 0, stream>>>(
      hb, wqkvB, nullptr, qkv, NROWS, 3*HIDDEN, HIDDEN);

  // 4. xPos rotary + head split
  k_rotary<<<BATCH*SEQ*NHEADS, 64, 0, stream>>>(qkv, qb, kb, vtb);

  // 5. retention (flash-style, WMMA both matmuls, async K/V streaming)
  k_retention<<<BATCH*NHEADS*(SEQ/ST), 128, 0, stream>>>(qb, kb, vtb, retb);

  // 6. groupnorm per (b,h), transpose to (B,S,HIDDEN)
  k_groupnorm<<<BATCH*NHEADS, 256, 0, stream>>>(retb, gn_w, gn_b, retn);

  // 7. gate: g = h @ Wg ; y = mish(g) * retn -> bf16
  k_gemm_bf16<<<dim3(HIDDEN/BN, NROWS/BM), 256, 0, stream>>>(
      hb, wgB, nullptr, gbuf, NROWS, HIDDEN, HIDDEN);
  k_mish_mul<<<EWG, 256, 0, stream>>>(gbuf, retn, yb, N);

  // 8. y2 = y @ Wo + x
  k_gemm_bf16<<<dim3(HIDDEN/BN, NROWS/BM), 256, 0, stream>>>(
      yb, woB, nullptr, y2, NROWS, HIDDEN, HIDDEN);
  k_residual_inplace<<<EWG, 256, 0, stream>>>(y2, x, N);

  // 9. z = LN2(y2) -> bf16
  k_layernorm_bf16<<<NROWS, 256, 0, stream>>>(y2, ln2_w, ln2_b, zb);

  // 10. FFN
  k_gemm_bf16<<<dim3(HIDDEN/BN, NROWS/BM), 256, 0, stream>>>(
      zb, w1B, w1_b, f1, NROWS, HIDDEN, HIDDEN);
  k_gelu_bf16<<<EWG, 256, 0, stream>>>(f1, f1g, N);
  k_gemm_bf16<<<dim3(HIDDEN/BN, NROWS/BM), 256, 0, stream>>>(
      f1g, w2B, w2_b, f2, NROWS, HIDDEN, HIDDEN);

  // 11. out = ffn + y2
  k_add_out<<<EWG, 256, 0, stream>>>(f2, y2, out, N);

  (void)in_sizes; (void)n_in; (void)out_size; (void)ws_size;
}
